// Attention_50732153700378
// MI455X (gfx1250) — matrix operand order repaired and checked by reading, not compiled, not run on hardware
//
#include <hip/hip_runtime.h>

typedef __attribute__((ext_vector_type(16))) __bf16 v16bf;
typedef __attribute__((ext_vector_type(8)))  float  v8f;

#define SB   2
#define SS   2048
#define HH   896
#define SNH  14
#define SNKV 2
#define SHD  64

static __device__ __forceinline__ v8f wmma_bf16(v16bf a, v16bf b, v8f c) {
  // 8 args: (neg_a, A, neg_b, B, c_mod, C, reuse_a, reuse_b)
  return __builtin_amdgcn_wmma_f32_16x16x32_bf16(false, a, false, b, (short)0, c, false, false);
}

// A-matrix 16x32 bf16 fragment from a row-major fp32 tile (row stride ld).
// Layout: lane&15 = row M; lanes 0-15 carry K=0..15, lanes 16-31 carry K=16..31.
static __device__ __forceinline__ v16bf loadA_tile(const float* base, int ld) {
  const int lane = threadIdx.x & 31;
  const float* r = base + (lane & 15) * ld + ((lane >> 4) << 4);
  const float4 f0 = *(const float4*)(r + 0);
  const float4 f1 = *(const float4*)(r + 4);
  const float4 f2 = *(const float4*)(r + 8);
  const float4 f3 = *(const float4*)(r + 12);
  v16bf a;
  a[0]  = (__bf16)f0.x; a[1]  = (__bf16)f0.y; a[2]  = (__bf16)f0.z; a[3]  = (__bf16)f0.w;
  a[4]  = (__bf16)f1.x; a[5]  = (__bf16)f1.y; a[6]  = (__bf16)f1.z; a[7]  = (__bf16)f1.w;
  a[8]  = (__bf16)f2.x; a[9]  = (__bf16)f2.y; a[10] = (__bf16)f2.z; a[11] = (__bf16)f2.w;
  a[12] = (__bf16)f3.x; a[13] = (__bf16)f3.y; a[14] = (__bf16)f3.z; a[15] = (__bf16)f3.w;
  return a;
}

// B-matrix 32(K)x16(N) fragment given B^T row-major (rows = N, contiguous K, stride ld).
// Layout: lane&15 = N column; lanes 0-15 carry K=0..15, lanes 16-31 carry K=16..31.
static __device__ __forceinline__ v16bf loadB_nt(const float* bt, int ld) {
  const int lane = threadIdx.x & 31;
  const float* r = bt + (lane & 15) * ld + ((lane >> 4) << 4);
  const float4 f0 = *(const float4*)(r + 0);
  const float4 f1 = *(const float4*)(r + 4);
  const float4 f2 = *(const float4*)(r + 8);
  const float4 f3 = *(const float4*)(r + 12);
  v16bf b;
  b[0]  = (__bf16)f0.x; b[1]  = (__bf16)f0.y; b[2]  = (__bf16)f0.z; b[3]  = (__bf16)f0.w;
  b[4]  = (__bf16)f1.x; b[5]  = (__bf16)f1.y; b[6]  = (__bf16)f1.z; b[7]  = (__bf16)f1.w;
  b[8]  = (__bf16)f2.x; b[9]  = (__bf16)f2.y; b[10] = (__bf16)f2.z; b[11] = (__bf16)f2.w;
  b[12] = (__bf16)f3.x; b[13] = (__bf16)f3.y; b[14] = (__bf16)f3.z; b[15] = (__bf16)f3.w;
  return b;
}

// B-matrix 32(K)x16(N) fragment from a K-major source: src[k*ld + n] (e.g. V[t, d]).
static __device__ __forceinline__ v16bf loadB_t(const float* src, int ld) {
  const int lane = threadIdx.x & 31;
  const float* c = src + ((lane >> 4) << 4) * ld + (lane & 15);
  v16bf b;
#pragma unroll
  for (int j = 0; j < 16; ++j) b[j] = (__bf16)c[j * ld];
  return b;
}

static __device__ __forceinline__ float redmax16(float x) {
  x = fmaxf(x, __shfl_xor(x, 1, 32));
  x = fmaxf(x, __shfl_xor(x, 2, 32));
  x = fmaxf(x, __shfl_xor(x, 4, 32));
  x = fmaxf(x, __shfl_xor(x, 8, 32));
  return x;
}
static __device__ __forceinline__ float redsum16(float x) {
  x += __shfl_xor(x, 1, 32);
  x += __shfl_xor(x, 2, 32);
  x += __shfl_xor(x, 4, 32);
  x += __shfl_xor(x, 8, 32);
  return x;
}

// ---- Stage 1: fused QKV projection (pre-RoPE). One wave per 16x16 output tile.
__global__ __launch_bounds__(32) void qkv_gemm_kernel(
    const float* __restrict__ hs,
    const float* __restrict__ Wq, const float* __restrict__ bq,
    const float* __restrict__ Wk, const float* __restrict__ bk,
    const float* __restrict__ Wv, const float* __restrict__ bv,
    float* __restrict__ Qr, float* __restrict__ Kr, float* __restrict__ Vr) {
  const int nt = blockIdx.x;   // 0..71  (56 Q tiles, 8 K tiles, 8 V tiles)
  const int mt = blockIdx.y;   // 0..255 (B*S/16)
  const int lane = threadIdx.x;
  const float* W; const float* bias; int n0; int kind;
  if (nt < 56)      { W = Wq; bias = bq; n0 = nt * 16;        kind = 0; }
  else if (nt < 64) { W = Wk; bias = bk; n0 = (nt - 56) * 16; kind = 1; }
  else              { W = Wv; bias = bv; n0 = (nt - 64) * 16; kind = 2; }

  v8f c = {};
  const float* Abase = hs + (long)mt * 16 * HH;
  const float* Bbase = W + (long)n0 * HH;
#pragma unroll 4
  for (int kk = 0; kk < HH; kk += 32) {
    c = wmma_bf16(loadA_tile(Abase + kk, HH), loadB_nt(Bbase + kk, HH), c);
  }

  const int col  = lane & 15;
  const int rowo = (lane >> 4) << 3;
  const int n = n0 + col;
  const float bval = bias[n];
  const int h = n >> 6, d = n & 63;
#pragma unroll
  for (int i = 0; i < 8; ++i) {
    const int m = mt * 16 + i + rowo;
    const int b_ = m >> 11;         // / S
    const int s  = m & (SS - 1);
    const float val = c[i] + bval;
    if (kind == 0)      Qr[(((long)b_ * SNH  + h) * SS + s) * SHD + d] = val;
    else if (kind == 1) Kr[(((long)b_ * SNKV + h) * SS + s) * SHD + d] = val;
    else                Vr[(((long)b_ * SNKV + h) * SS + s) * SHD + d] = val;
  }
}

// ---- Stage 2: in-place RoPE. Each thread owns the (d, d+32) pair -> no race.
__global__ __launch_bounds__(256) void rope_kernel(float* __restrict__ X,
    const float* __restrict__ cs, const float* __restrict__ sn, int total) {
  const int idx = blockIdx.x * 256 + threadIdx.x;
  if (idx >= total) return;
  const int d = idx & 31;
  const int s = (idx >> 5) & (SS - 1);
  const long bh = (long)(idx >> 16);
  float* p = X + (bh * SS + s) * SHD;
  const float x1 = p[d], x2 = p[d + 32];
  const float c0 = cs[s * SHD + d],      s0 = sn[s * SHD + d];
  const float c1 = cs[s * SHD + d + 32], s1 = sn[s * SHD + d + 32];
  p[d]      = x1 * c0 - x2 * s0;
  p[d + 32] = x2 * c1 + x1 * s1;
}

// ---- Stage 3: causal GQA attention. One wave per (b, h, 16-row query block).
// Pass 1: online max/sum over causal K tiles. Pass 2: recompute scores (K/V sit
// in L2), write normalized attn_weights exactly once, accumulate P@V via LDS
// staging of P (D-layout -> A-layout).
__global__ __launch_bounds__(32) void attn_kernel(
    const float* __restrict__ Qr, const float* __restrict__ Kr, const float* __restrict__ Vr,
    float* __restrict__ Wout, float* __restrict__ O) {
  __shared__ float Pld[16 * 32];
  const int mb = blockIdx.x;       // 0..127
  const int h  = blockIdx.y;       // 0..13
  const int b  = blockIdx.z;       // 0..1
  const int lane = threadIdx.x;
  const int kvh = h / 7;           // G = 7
  const int mbase = mb * 16;
  const int tend  = mbase + 16;    // causal extent (multiple of 16)
  const float* Qp = Qr + (((long)b * SNH + h) * SS + mbase) * SHD;
  const float* Kp = Kr + ((long)b * SNKV + kvh) * SS * SHD;
  const float* Vp = Vr + ((long)b * SNKV + kvh) * SS * SHD;

  const v16bf aq0 = loadA_tile(Qp, SHD);        // Q k = 0..31
  const v16bf aq1 = loadA_tile(Qp + 32, SHD);   // Q k = 32..63
  const int col  = lane & 15;
  const int rowo = (lane >> 4) << 3;

  float mrow[8], lrow[8];
#pragma unroll
  for (int i = 0; i < 8; ++i) { mrow[i] = -1e30f; lrow[i] = 0.0f; }

  // Pass 1: running row max / sum-exp
  for (int tb = 0; tb < tend; tb += 16) {
    v8f sc = {};
    sc = wmma_bf16(aq0, loadB_nt(Kp + (long)tb * SHD, SHD), sc);
    sc = wmma_bf16(aq1, loadB_nt(Kp + (long)tb * SHD + 32, SHD), sc);
    const int t = tb + col;
#pragma unroll
    for (int i = 0; i < 8; ++i) {
      const int sg = mbase + i + rowo;
      const float sv = (t <= sg) ? sc[i] * 0.125f : -1e30f;
      const float tm = redmax16(sv);
      const float mn = fmaxf(mrow[i], tm);
      const float e  = __expf(sv - mn);
      const float ts = redsum16(e);
      lrow[i] = lrow[i] * __expf(mrow[i] - mn) + ts;
      mrow[i] = mn;
    }
  }
  float invl[8];
#pragma unroll
  for (int i = 0; i < 8; ++i) invl[i] = 1.0f / lrow[i];

  // Pass 2: write normalized w once; accumulate O = w @ V
  v8f acc0 = {}, acc1 = {}, acc2 = {}, acc3 = {};
  const long wb = (((long)b * SNH + h) * SS + mbase) * SS;

  for (int tb = 0; tb < tend; tb += 32) {
#pragma unroll
    for (int sub = 0; sub < 2; ++sub) {
      const int t0 = tb + sub * 16;
      v8f sc = {};
      sc = wmma_bf16(aq0, loadB_nt(Kp + (long)t0 * SHD, SHD), sc);
      sc = wmma_bf16(aq1, loadB_nt(Kp + (long)t0 * SHD + 32, SHD), sc);
      const int t = t0 + col;
#pragma unroll
      for (int i = 0; i < 8; ++i) {
        const int sg = mbase + i + rowo;
        const float sv = (t <= sg) ? sc[i] * 0.125f : -1e30f;
        const float w  = __expf(sv - mrow[i]) * invl[i];
        Pld[(i + rowo) * 32 + sub * 16 + col] = w;
        Wout[wb + (long)(i + rowo) * SS + t] = w;
      }
    }
    __syncthreads();
    const v16bf ap = loadA_tile(Pld, 32);   // P 16x32 as A fragment
    acc0 = wmma_bf16(ap, loadB_t(Vp + (long)tb * SHD + 0,  SHD), acc0);
    acc1 = wmma_bf16(ap, loadB_t(Vp + (long)tb * SHD + 16, SHD), acc1);
    acc2 = wmma_bf16(ap, loadB_t(Vp + (long)tb * SHD + 32, SHD), acc2);
    acc3 = wmma_bf16(ap, loadB_t(Vp + (long)tb * SHD + 48, SHD), acc3);
    __syncthreads();
  }

  // Zero-fill the untouched (strictly future) attn_weights columns.
  const int tcov = (tend + 31) & ~31;
  const int rem  = SS - tcov;
  if (rem > 0) {
    const int rem4 = rem >> 2;
    const float4 z4 = make_float4(0.f, 0.f, 0.f, 0.f);
    for (int i = lane; i < 16 * rem4; i += 32) {
      const int r  = i / rem4;
      const int c4 = i - r * rem4;
      *(float4*)(Wout + wb + (long)r * SS + tcov + 4 * c4) = z4;
    }
  }

  // Store O block in [B, S, NH, HD] (== row-major [B*S, 896] for the O-proj GEMM).
#pragma unroll
  for (int i = 0; i < 8; ++i) {
    const int sg = mbase + i + rowo;
    float* ob = O + (((long)b * SS + sg) * SNH + h) * SHD;
    ob[col]      = acc0[i];
    ob[16 + col] = acc1[i];
    ob[32 + col] = acc2[i];
    ob[48 + col] = acc3[i];
  }
}

// ---- Stage 4: output projection out = O @ Wo^T
__global__ __launch_bounds__(32) void oproj_kernel(const float* __restrict__ Oin,
    const float* __restrict__ Wo, float* __restrict__ out) {
  const int nt = blockIdx.x;   // 0..55
  const int mt = blockIdx.y;   // 0..255
  const int n0 = nt * 16;
  v8f c = {};
  const float* Abase = Oin + (long)mt * 16 * HH;
  const float* Bbase = Wo + (long)n0 * HH;
#pragma unroll 4
  for (int kk = 0; kk < HH; kk += 32) {
    c = wmma_bf16(loadA_tile(Abase + kk, HH), loadB_nt(Bbase + kk, HH), c);
  }
  const int lane = threadIdx.x;
  const int col  = lane & 15;
  const int rowo = (lane >> 4) << 3;
#pragma unroll
  for (int i = 0; i < 8; ++i) {
    const long m = (long)mt * 16 + i + rowo;
    out[m * HH + n0 + col] = c[i];
  }
}

extern "C" void kernel_launch(void* const* d_in, const int* in_sizes, int n_in,
                              void* d_out, int out_size, void* d_ws, size_t ws_size,
                              hipStream_t stream) {
  (void)in_sizes; (void)n_in; (void)out_size; (void)ws_size;
  const float* hs   = (const float*)d_in[0];
  const float* cosp = (const float*)d_in[1];
  const float* sinp = (const float*)d_in[2];
  const float* Wq   = (const float*)d_in[3];
  const float* bq   = (const float*)d_in[4];
  const float* Wk   = (const float*)d_in[5];
  const float* bk   = (const float*)d_in[6];
  const float* Wv   = (const float*)d_in[7];
  const float* bv   = (const float*)d_in[8];
  const float* Wo   = (const float*)d_in[9];
  // d_in[10] = attention_mask (causal; computed analytically in-kernel)

  float* Qr = (float*)d_ws;                               // [B,NH,S,HD]
  float* Kr = Qr + (size_t)SB * SNH  * SS * SHD;          // [B,NKV,S,HD]
  float* Vr = Kr + (size_t)SB * SNKV * SS * SHD;          // [B,NKV,S,HD]
  float* O  = Vr + (size_t)SB * SNKV * SS * SHD;          // [B,S,NH,HD]

  float* outp = (float*)d_out;                            // attn_output [B,S,H]
  float* wout = outp + (size_t)SB * SS * HH;              // attn_weights [B,NH,S,S]

  qkv_gemm_kernel<<<dim3(72, 256), 32, 0, stream>>>(hs, Wq, bq, Wk, bk, Wv, bv, Qr, Kr, Vr);
  rope_kernel<<<(SB * SNH  * SS * 32) / 256, 256, 0, stream>>>(Qr, cosp, sinp, SB * SNH  * SS * 32);
  rope_kernel<<<(SB * SNKV * SS * 32) / 256, 256, 0, stream>>>(Kr, cosp, sinp, SB * SNKV * SS * 32);
  attn_kernel<<<dim3(SS / 16, SNH, SB), 32, 0, stream>>>(Qr, Kr, Vr, wout, O);
  oproj_kernel<<<dim3(56, 256), 32, 0, stream>>>(O, Wo, outp);
}
